// DeformableCrossAttention_6459630813219
// MI455X (gfx1250) — matrix operand order repaired and checked by reading, not compiled, not run on hardware
//
#include <hip/hip_runtime.h>
#include <hip/hip_bf16.h>

// ---------------------------------------------------------------------------
// Deformable cross-attention on gfx1250 (CDNA5, wave32, WMMA).
//   1) cvt x, context -> f16
//   2) transpose-cvt Wq, Woff, Watt -> [N,K] f16
//   3) fuse Wkv_t=(Wk@Wv)^T, Wov_t=(Wo@Wout)^T f16, bov = bo@Wout+bout
//   4) WMMA: q = x@Wq_t (f16);  val = ctx@Wkv_t+bv (f16)
//   5) WMMA: off = q@Woff_t+boff (f32, N=64); att = q@Watt_t+batt (f32, N=32)
//   6) softmax4; deformable bilinear gather -> samp f16
//   7) WMMA: d_out = samp@Wov_t + bov (f32)
// GEMM kernel is fully straight-line in the K-loop (TILES/ROWS templated) so
// accumulators stay pinned in VGPRs across v_wmma ops.
// ---------------------------------------------------------------------------

typedef _Float16 f16;
typedef _Float16 v16h __attribute__((ext_vector_type(16)));
typedef _Float16 v8h  __attribute__((ext_vector_type(8)));
typedef float    v8f  __attribute__((ext_vector_type(8)));

union Frag16 {
  v16h v;
  v8h  h[2];
};

// ---------------- elementwise f32 -> f16 ----------------
__global__ void k_cvt_f16(const float* __restrict__ s, f16* __restrict__ d, int n) {
  int i = blockIdx.x * blockDim.x + threadIdx.x;
  if (i < n) d[i] = (f16)s[i];
}

// ---------------- W[K,N] f32 -> Wt[N,K] f16 ----------------
__global__ void k_transpose_cvt(const float* __restrict__ W, f16* __restrict__ Wt,
                                int K, int N) {
  int i = blockIdx.x * blockDim.x + threadIdx.x;
  if (i >= K * N) return;
  int n = i / K, k = i - n * K;
  Wt[i] = (f16)W[(size_t)k * N + n];
}

// ---------------- Ct[N,K] f16 = (A[K,J] @ B[J,N])^T ----------------
__global__ void k_fuse_weights_t(const float* __restrict__ A, const float* __restrict__ B,
                                 f16* __restrict__ Ct, int K, int J, int N) {
  int i = blockIdx.x * blockDim.x + threadIdx.x;
  if (i >= K * N) return;
  int n = i / K, k = i - n * K;
  float s = 0.f;
  for (int j = 0; j < J; ++j) s += A[(size_t)k * J + j] * B[(size_t)j * N + n];
  Ct[i] = (f16)s;
}

// ---------------- out[n] = b1 @ W2 + b2 ----------------
__global__ void k_fuse_bias(const float* __restrict__ b1, const float* __restrict__ W2,
                            const float* __restrict__ b2, float* __restrict__ out,
                            int J, int N) {
  int n = blockIdx.x * blockDim.x + threadIdx.x;
  if (n >= N) return;
  float s = b2[n];
  for (int j = 0; j < J; ++j) s += b1[j] * W2[(size_t)j * N + n];
  out[n] = s;
}

// ---------------- WMMA GEMM: C[M,N] = A[M,K] @ Bt[N,K]^T (+bias) ----------------
// block = 256 threads = 8 waves. Each wave: ROWS 16-row strips (128 apart)
// x TILES 16-col tiles. Requires M % (128*ROWS) == 0, N == gridDim.y*TILES*16,
// K % 32 == 0. No control flow in the K-loop: EXEC all-ones, acc pinned.
template <int ROWS, int TILES, bool OUT_F16>
__global__ void __launch_bounds__(256)
k_gemm_wmma(const f16* __restrict__ A, const f16* __restrict__ Bt,
            const float* __restrict__ bias, void* __restrict__ Cout,
            int N, int K) {
  const int lane = threadIdx.x & 31;
  const int wave = threadIdx.x >> 5;
  const int rowBase = blockIdx.x * (128 * ROWS) + wave * 16;
  const int colBase = blockIdx.y * (TILES * 16);
  const int l16 = lane & 15;
  const int sel = lane >> 4;  // half-wave select

  v8f z = {0.f, 0.f, 0.f, 0.f, 0.f, 0.f, 0.f, 0.f};
  v8f acc[ROWS][TILES];
#pragma unroll
  for (int r = 0; r < ROWS; ++r)
#pragma unroll
    for (int t = 0; t < TILES; ++t) acc[r][t] = z;

  // A fragment (16x32 f16): lane<16 holds K kb..kb+15;
  // lane>=16 K kb+16..kb+31 -> two b128 loads into one Frag16.
  const f16* arow[ROWS];
#pragma unroll
  for (int r = 0; r < ROWS; ++r)
    arow[r] = A + (size_t)(rowBase + r * 128 + l16) * K + 16 * sel;

  // B fragment (32x16 f16): lane = column; lanes 0-15 K=kb..kb+15,
  // lanes 16-31 K=kb+16..kb+31 -> one contiguous 32B v16h load.
  const f16* brow[TILES];
#pragma unroll
  for (int t = 0; t < TILES; ++t)
    brow[t] = Bt + (size_t)(colBase + t * 16 + l16) * K + 16 * sel;

  for (int kb = 0; kb < K; kb += 32) {
    Frag16 af[ROWS];
#pragma unroll
    for (int r = 0; r < ROWS; ++r) {
      af[r].h[0] = *(const v8h*)(arow[r] + kb);
      af[r].h[1] = *(const v8h*)(arow[r] + kb + 8);
    }
#pragma unroll
    for (int t = 0; t < TILES; ++t) {
      v16h bf = *(const v16h*)(brow[t] + kb);
#pragma unroll
      for (int r = 0; r < ROWS; ++r) {
        acc[r][t] = __builtin_amdgcn_wmma_f32_16x16x32_f16(
            false, af[r].v, false, bf, (short)0, acc[r][t], false, false);
      }
    }
  }

  // C/D layout: lane<16 -> n=l16, VGPR g holds M=rowBase+g (+8 for lane>=16)
#pragma unroll
  for (int r = 0; r < ROWS; ++r) {
#pragma unroll
    for (int t = 0; t < TILES; ++t) {
      int ncol = colBase + t * 16 + l16;
      float badd = bias ? bias[ncol] : 0.f;
#pragma unroll
      for (int g = 0; g < 8; ++g) {
        int row = rowBase + r * 128 + g + 8 * sel;
        float v = acc[r][t][g] + badd;
        if (OUT_F16) ((f16*)Cout)[(size_t)row * N + ncol] = (f16)v;
        else         ((float*)Cout)[(size_t)row * N + ncol] = v;
      }
    }
  }
}

// ---------------- softmax over groups of 4 (L*P = 4) ----------------
__global__ void k_softmax4(float* __restrict__ a, int ngroups) {
  int i = blockIdx.x * blockDim.x + threadIdx.x;
  if (i >= ngroups) return;
  float* p = a + (size_t)i * 4;
  float a0 = p[0], a1 = p[1], a2 = p[2], a3 = p[3];
  float mx = fmaxf(fmaxf(a0, a1), fmaxf(a2, a3));
  float e0 = __expf(a0 - mx), e1 = __expf(a1 - mx);
  float e2 = __expf(a2 - mx), e3 = __expf(a3 - mx);
  float r = 1.f / (e0 + e1 + e2 + e3);
  p[0] = e0 * r; p[1] = e1 * r; p[2] = e2 * r; p[3] = e3 * r;
}

// ---------------- deformable bilinear gather ----------------
// One block per (b,q): threadIdx = head(8) x channel-pair(32). H=W=64, D=64.
__global__ void __launch_bounds__(256)
k_deform(const f16* __restrict__ value, const float* __restrict__ off,
         const float* __restrict__ attw, f16* __restrict__ samp) {
  const int bq = blockIdx.x;             // b*4096 + q
  const int q  = bq & 4095;
  const int b  = bq >> 12;
  const int m  = threadIdx.x >> 5;       // head
  const int d2 = threadIdx.x & 31;       // channel pair
  const int qy = q >> 6, qx = q & 63;
  const float refx = (qx + 0.5f) * (1.f / 64.f);
  const float refy = (qy + 0.5f) * (1.f / 64.f);
  const float* op = off  + ((size_t)bq * 8 + m) * 8;   // [P=4][2]
  const float* ap = attw + ((size_t)bq * 8 + m) * 4;   // [P=4]
  const f16* vbase = value + (size_t)b * 4096 * 512 + m * 64 + d2 * 2;

  float o0 = 0.f, o1 = 0.f;
#pragma unroll
  for (int p = 0; p < 4; ++p) {
    // loc = ref + off/(W,H); pixel = loc*W - 0.5 (grid_sample align_corners=False)
    float x = (refx + op[2 * p]     * (1.f / 64.f)) * 64.f - 0.5f;
    float y = (refy + op[2 * p + 1] * (1.f / 64.f)) * 64.f - 0.5f;
    float x0f = floorf(x), y0f = floorf(y);
    float wx1 = x - x0f, wy1 = y - y0f;
    int x0 = (int)x0f, y0 = (int)y0f;
    float aw = ap[p];
#pragma unroll
    for (int dy = 0; dy < 2; ++dy) {
#pragma unroll
      for (int dx = 0; dx < 2; ++dx) {
        int xi = x0 + dx, yi = y0 + dy;
        if (xi >= 0 && xi < 64 && yi >= 0 && yi < 64) {   // zeros padding
          float w = (dx ? wx1 : 1.f - wx1) * (dy ? wy1 : 1.f - wy1) * aw;
          const f16* vp = vbase + (size_t)(yi * 64 + xi) * 512;
          o0 += w * (float)vp[0];
          o1 += w * (float)vp[1];
        }
      }
    }
  }
  f16* sp = samp + (size_t)bq * 512 + m * 64 + d2 * 2;
  sp[0] = (f16)o0;
  sp[1] = (f16)o1;
}

// ---------------------------------------------------------------------------
extern "C" void kernel_launch(void* const* d_in, const int* in_sizes, int n_in,
                              void* d_out, int out_size, void* d_ws, size_t ws_size,
                              hipStream_t stream) {
  (void)in_sizes; (void)n_in; (void)out_size; (void)ws_size;
  const float* x    = (const float*)d_in[0];
  const float* ctx  = (const float*)d_in[1];
  // d_in[2], d_in[3]: spatial shapes (static 64x64, hardcoded)
  const float* Wq   = (const float*)d_in[4];
  const float* Wk   = (const float*)d_in[5];
  const float* Wv   = (const float*)d_in[6];
  const float* bv   = (const float*)d_in[7];
  const float* Woff = (const float*)d_in[8];
  const float* boff = (const float*)d_in[9];
  const float* Watt = (const float*)d_in[10];
  const float* batt = (const float*)d_in[11];
  const float* Wo   = (const float*)d_in[12];
  const float* bo   = (const float*)d_in[13];
  const float* Wout = (const float*)d_in[14];
  const float* bout = (const float*)d_in[15];

  const int NTOK = 8 * 4096;  // 32768 tokens
  const int DIN = 512;
  const int nElem = NTOK * DIN;

  // ---- workspace carve-out (256B aligned) ----
  size_t cur = 0;
  char* ws = (char*)d_ws;
  auto alloc = [&](size_t bytes) -> char* {
    cur = (cur + 255) & ~(size_t)255;
    char* p = ws + cur;
    cur += bytes;
    return p;
  };
  f16*   x_h    = (f16*)alloc((size_t)nElem * 2);
  f16*   ctx_h  = (f16*)alloc((size_t)nElem * 2);
  f16*   q_h    = (f16*)alloc((size_t)nElem * 2);
  f16*   val_h  = (f16*)alloc((size_t)nElem * 2);
  f16*   samp_h = (f16*)alloc((size_t)nElem * 2);
  float* off_f  = (float*)alloc((size_t)NTOK * 64 * 4);
  float* att_f  = (float*)alloc((size_t)NTOK * 32 * 4);
  f16*   Wq_t   = (f16*)alloc(512 * 512 * 2);
  f16*   Wkv_t  = (f16*)alloc(512 * 512 * 2);
  f16*   Wov_t  = (f16*)alloc(512 * 512 * 2);
  f16*   Woff_t = (f16*)alloc(512 * 64 * 2);
  f16*   Watt_t = (f16*)alloc(512 * 32 * 2);
  float* bov    = (float*)alloc(512 * 4);

  // ---- 1) activation conversion ----
  k_cvt_f16<<<nElem / 256, 256, 0, stream>>>(x, x_h, nElem);
  k_cvt_f16<<<nElem / 256, 256, 0, stream>>>(ctx, ctx_h, nElem);

  // ---- 2) weight transpose-conversion ----
  k_transpose_cvt<<<(512 * 512) / 256, 256, 0, stream>>>(Wq, Wq_t, 512, 512);
  k_transpose_cvt<<<(512 * 64) / 256, 256, 0, stream>>>(Woff, Woff_t, 512, 64);
  k_transpose_cvt<<<(512 * 32) / 256, 256, 0, stream>>>(Watt, Watt_t, 512, 32);

  // ---- 3) algebraic fusion ----
  k_fuse_weights_t<<<(512 * 512) / 256, 256, 0, stream>>>(Wk, Wv, Wkv_t, 512, 512, 512);
  k_fuse_weights_t<<<(512 * 512) / 256, 256, 0, stream>>>(Wo, Wout, Wov_t, 512, 512, 512);
  k_fuse_bias<<<2, 256, 0, stream>>>(bo, Wout, bout, bov, 512, 512);

  // ---- 4-5) WMMA GEMMs (ROWS=2 -> 256 rows/block) ----
  dim3 blk(256);
  dim3 g512(NTOK / 256, 512 / 64);   // TILES=4
  dim3 g1(NTOK / 256, 1);
  k_gemm_wmma<2, 4, true><<<g512, blk, 0, stream>>>(x_h, Wq_t, nullptr, q_h, 512, 512);
  k_gemm_wmma<2, 4, true><<<g512, blk, 0, stream>>>(ctx_h, Wkv_t, bv, val_h, 512, 512);
  k_gemm_wmma<2, 4, false><<<g1, blk, 0, stream>>>(q_h, Woff_t, boff, off_f, 64, 512);
  k_gemm_wmma<2, 2, false><<<g1, blk, 0, stream>>>(q_h, Watt_t, batt, att_f, 32, 512);
  k_softmax4<<<(NTOK * 8) / 256, 256, 0, stream>>>(att_f, NTOK * 8);

  // ---- 6) deformable gather ----
  k_deform<<<NTOK, 256, 0, stream>>>(val_h, off_f, att_f, samp_h);

  // ---- 7) fused output projection -> d_out (f32) ----
  k_gemm_wmma<2, 4, false><<<g512, blk, 0, stream>>>(samp_h, Wov_t, bov, (float*)d_out,
                                                     512, 512);
}
